// CrossPath_37426345018111
// MI455X (gfx1250) — matrix operand order repaired and checked by reading, not compiled, not run on hardware
//
#include <hip/hip_runtime.h>

#define B_    32
#define N_    1024
#define C_    256
#define C2_   512
#define H_    8
#define CLS_  6
#define BN_   (B_ * N_)
#define SCALE_ 0.17677669529663687f
#define EPS_   1e-5f

typedef __attribute__((ext_vector_type(16))) _Float16 v16h;
typedef __attribute__((ext_vector_type(8)))  _Float16 v8h;
typedef __attribute__((ext_vector_type(8)))  float    v8f;
typedef int vint4 __attribute__((__vector_size__(4 * sizeof(int))));

#if defined(__has_builtin)
#if __has_builtin(__builtin_amdgcn_global_load_async_to_lds_b128)
#define CTX_ASYNC 1
#endif
#endif
#ifndef CTX_ASYNC
#define CTX_ASYNC 0
#endif

#if defined(__has_builtin) && __has_builtin(__builtin_amdgcn_s_wait_asynccnt)
#define WAIT_ASYNCCNT(n) __builtin_amdgcn_s_wait_asynccnt(n)
#else
#define WAIT_ASYNCCNT(n) asm volatile("s_wait_asynccnt %0" ::"n"(n) : "memory")
#endif

static __device__ __forceinline__ v8f wmma_f16(v16h a, v16h b, v8f c) {
  // D = A(16x32) * B(32x16) + C, f32 accumulate
  return __builtin_amdgcn_wmma_f32_16x16x32_f16(false, a, false, b, (short)0, c,
                                                false, false);
}

static __device__ __forceinline__ v16h pack2(v8h lo, v8h hi) {
  v16h r;
#pragma unroll
  for (int i = 0; i < 8; ++i) { r[i] = lo[i]; r[8 + i] = hi[i]; }
  return r;
}

// f16 A/B fragment: two contiguous 8-half chunks at +0 and +8 halfs
static __device__ __forceinline__ v16h load_frag_h(const _Float16* p) {
  return pack2(*(const v8h*)p, *(const v8h*)(p + 8));
}

// fp32 A fragment converted in-register to f16
static __device__ __forceinline__ v16h cvt_frag_f(const float* p) {
  v8f a0 = *(const v8f*)p;
  v8f a1 = *(const v8f*)(p + 8);
  v16h a;
#pragma unroll
  for (int i = 0; i < 8; ++i) { a[i] = (_Float16)a0[i]; a[8 + i] = (_Float16)a1[i]; }
  return a;
}

#if CTX_ASYNC
// Async DMA one 16B chunk (per lane) from global into LDS (ASYNCcnt-tracked).
// Builtin signature (from compiler diagnostic): param type is generic
// `int __vector(4) *` — pass plain vint4* pointers.
static __device__ __forceinline__ void async_cp16(const _Float16* g, _Float16* l) {
  __builtin_amdgcn_global_load_async_to_lds_b128((vint4*)g, (vint4*)l, 0, 0);
}
#endif

// ---------------------------------------------------------------------------
// Weight prep: fp32 [K][Nc] row-major -> f16 [Nc][K].
// ---------------------------------------------------------------------------
__global__ void transpose_w_kernel(const float* __restrict__ Win,
                                   _Float16* __restrict__ Wout, int K, int Nc) {
  int idx = blockIdx.x * blockDim.x + threadIdx.x;
  if (idx >= K * Nc) return;
  int n = idx / K;
  int k = idx - n * K;
  Wout[idx] = (_Float16)Win[(size_t)k * Nc + n];
}

// ---------------------------------------------------------------------------
// proj: Out[BN,512] = f16( relu( X[BN,256] @ W + bias ) ), Wt = f16 [512][256]
// 32x32 tile per wave (2x2 register blocking): 4 WMMAs / K-step, 32 total.
// ---------------------------------------------------------------------------
__global__ void proj_gemm_kernel(const float* __restrict__ X,
                                 const _Float16* __restrict__ Wt,
                                 const float* __restrict__ bias,
                                 _Float16* __restrict__ Out) {
  const int lane = threadIdx.x & 31;
  const int wave = threadIdx.x >> 5;
  const int tile = blockIdx.x * 8 + wave;        // 32x32 tiles, 16 col-tiles
  const int row0 = (tile >> 4) << 5;
  const int col0 = (tile & 15) << 5;
  const int m  = lane & 15;
  const int kh = lane >> 4;
  const float*    a0p = X  + (size_t)(row0 + m) * C_ + kh * 16;
  const float*    a1p = a0p + (size_t)16 * C_;
  const _Float16* b0p = Wt + (size_t)(col0 + m) * C_ + kh * 16;
  const _Float16* b1p = b0p + (size_t)16 * C_;
  v8f c00 = {}, c01 = {}, c10 = {}, c11 = {};
#pragma unroll
  for (int kb = 0; kb < C_; kb += 32) {
    v16h a0 = cvt_frag_f(a0p + kb);
    v16h a1 = cvt_frag_f(a1p + kb);
    v16h b0 = *(const v16h*)(b0p + kb);
    v16h b1 = *(const v16h*)(b1p + kb);
    c00 = wmma_f16(a0, b0, c00);
    c01 = wmma_f16(a0, b1, c01);
    c10 = wmma_f16(a1, b0, c10);
    c11 = wmma_f16(a1, b1, c11);
  }
  const int rb = row0 + 8 * kh;
  {
    const int col = col0 + m;
    const float bv = bias[col];
#pragma unroll
    for (int r = 0; r < 8; ++r) {
      float v0 = c00[r] + bv; v0 = v0 > 0.f ? v0 : 0.f;
      Out[(size_t)(rb + r) * C2_ + col] = (_Float16)v0;
      float v1 = c10[r] + bv; v1 = v1 > 0.f ? v1 : 0.f;
      Out[(size_t)(rb + 16 + r) * C2_ + col] = (_Float16)v1;
    }
  }
  {
    const int col = col0 + 16 + m;
    const float bv = bias[col];
#pragma unroll
    for (int r = 0; r < 8; ++r) {
      float v0 = c01[r] + bv; v0 = v0 > 0.f ? v0 : 0.f;
      Out[(size_t)(rb + r) * C2_ + col] = (_Float16)v0;
      float v1 = c11[r] + bv; v1 = v1 > 0.f ? v1 : 0.f;
      Out[(size_t)(rb + 16 + r) * C2_ + col] = (_Float16)v1;
    }
  }
}

// ---------------------------------------------------------------------------
// kv: Out[BN,512] = f16( A[BN,256-window] @ Wkv ), A row stride 512.
// 32x32 tile per wave, 32 WMMAs.
// ---------------------------------------------------------------------------
__global__ void kv_gemm_kernel(const _Float16* __restrict__ A,
                               const _Float16* __restrict__ Wt,
                               _Float16* __restrict__ Out) {
  const int lane = threadIdx.x & 31;
  const int wave = threadIdx.x >> 5;
  const int tile = blockIdx.x * 8 + wave;
  const int row0 = (tile >> 4) << 5;
  const int col0 = (tile & 15) << 5;
  const int m  = lane & 15;
  const int kh = lane >> 4;
  const _Float16* a0p = A  + (size_t)(row0 + m) * C2_ + kh * 16;
  const _Float16* a1p = a0p + (size_t)16 * C2_;
  const _Float16* b0p = Wt + (size_t)(col0 + m) * C_ + kh * 16;
  const _Float16* b1p = b0p + (size_t)16 * C_;
  v8f c00 = {}, c01 = {}, c10 = {}, c11 = {};
#pragma unroll
  for (int kb = 0; kb < C_; kb += 32) {
    v16h a0 = load_frag_h(a0p + kb);
    v16h a1 = load_frag_h(a1p + kb);
    v16h b0 = *(const v16h*)(b0p + kb);
    v16h b1 = *(const v16h*)(b1p + kb);
    c00 = wmma_f16(a0, b0, c00);
    c01 = wmma_f16(a0, b1, c01);
    c10 = wmma_f16(a1, b0, c10);
    c11 = wmma_f16(a1, b1, c11);
  }
  const int rb = row0 + 8 * kh;
  {
    const int col = col0 + m;
#pragma unroll
    for (int r = 0; r < 8; ++r) {
      Out[(size_t)(rb + r) * C2_ + col]      = (_Float16)c00[r];
      Out[(size_t)(rb + 16 + r) * C2_ + col] = (_Float16)c10[r];
    }
  }
  {
    const int col = col0 + 16 + m;
#pragma unroll
    for (int r = 0; r < 8; ++r) {
      Out[(size_t)(rb + r) * C2_ + col]      = (_Float16)c01[r];
      Out[(size_t)(rb + 16 + r) * C2_ + col] = (_Float16)c11[r];
    }
  }
}

// ---------------------------------------------------------------------------
// ctx: per (b,h): S[d,e] = SCALE * sum_n k[n,d]*v[n,e]; softmax over d;
// output ctxT f16 [e][d]. Block = 128 threads (4 waves = 2x2 16x16 tiles).
// Tiles of 128 n staged into LDS; with CTX_ASYNC, tiles are double-buffered
// via GLOBAL_LOAD_ASYNC_TO_LDS_B128 + s_wait_asynccnt, overlapping the DMA of
// tile nt+1 with the transpose+WMMA of tile nt.
// ---------------------------------------------------------------------------
__global__ void ctx_kernel(const _Float16* __restrict__ KV,
                           _Float16* __restrict__ CtxT) {
#if CTX_ASYNC
  __shared__ alignas(16) _Float16 stage[2][128][64];   // [buf][n][k32|v32]
#endif
  __shared__ alignas(32) _Float16 kT[32][144];         // [d][n] transposed
  __shared__ alignas(32) _Float16 vT[32][144];         // [e][n] transposed
  __shared__ float S[32][33];
  const int bh = blockIdx.x;
  const int b = bh >> 3, h = bh & 7;
  const int t = threadIdx.x;               // 0..127
  const int lane = t & 31, wave = t >> 5;
  const int dt = wave >> 1, et = wave & 1;
  const int m = lane & 15, kh = lane >> 4;
  const int mrow = dt * 16 + m;
  const int erow = et * 16 + m;
  const _Float16* kp0 = KV + (size_t)(b * N_ + t) * C2_ + h * 32;
  v8f c = {};
#if CTX_ASYNC
  {
    const _Float16* kp = kp0;
#pragma unroll
    for (int i = 0; i < 4; ++i) {
      async_cp16(kp + i * 8,      &stage[0][t][i * 8]);
      async_cp16(kp + C_ + i * 8, &stage[0][t][32 + i * 8]);
    }
  }
#endif
  for (int nt = 0; nt < 8; ++nt) {
#if CTX_ASYNC
    const int cur = nt & 1;
    if (nt < 7) {
      const _Float16* kp = kp0 + (size_t)(nt + 1) * 128 * C2_;
#pragma unroll
      for (int i = 0; i < 4; ++i) {
        async_cp16(kp + i * 8,      &stage[cur ^ 1][t][i * 8]);
        async_cp16(kp + C_ + i * 8, &stage[cur ^ 1][t][32 + i * 8]);
      }
      WAIT_ASYNCCNT(8);       // current tile's 8 copies complete
    } else {
      WAIT_ASYNCCNT(0);
    }
    // Each thread transposes its own staged row (only producer/consumer).
#pragma unroll
    for (int i = 0; i < 4; ++i) {
      v8h ka = *(const v8h*)&stage[cur][t][i * 8];
      v8h va = *(const v8h*)&stage[cur][t][32 + i * 8];
#pragma unroll
      for (int j = 0; j < 8; ++j) { kT[i * 8 + j][t] = ka[j]; vT[i * 8 + j][t] = va[j]; }
    }
#else
    const _Float16* kp = kp0 + (size_t)nt * 128 * C2_;
#pragma unroll
    for (int i = 0; i < 4; ++i) {
      v8h ka = *(const v8h*)(kp + i * 8);
      v8h va = *(const v8h*)(kp + C_ + i * 8);
#pragma unroll
      for (int j = 0; j < 8; ++j) { kT[i * 8 + j][t] = ka[j]; vT[i * 8 + j][t] = va[j]; }
    }
#endif
    __syncthreads();
#pragma unroll
    for (int kb = 0; kb < 128; kb += 32) {
      v16h a  = pack2(*(const v8h*)&kT[mrow][kb + kh * 16],
                      *(const v8h*)&kT[mrow][kb + kh * 16 + 8]);
      v16h bf = *(const v16h*)&vT[erow][kb + kh * 16];
      c = wmma_f16(a, bf, c);
    }
    __syncthreads();
  }
#pragma unroll
  for (int r = 0; r < 8; ++r)
    S[dt * 16 + 8 * kh + r][et * 16 + m] = c[r] * SCALE_;
  __syncthreads();
  if (t < 32) {
    const int e = t;
    float mx = -1e30f;
#pragma unroll
    for (int d = 0; d < 32; ++d) mx = fmaxf(mx, S[d][e]);
    float ex[32];
    float sum = 0.f;
#pragma unroll
    for (int d = 0; d < 32; ++d) { ex[d] = expf(S[d][e] - mx); sum += ex[d]; }
    float inv = 1.f / sum;
    _Float16* out = CtxT + (size_t)bh * 1024 + e * 32;
#pragma unroll
    for (int d = 0; d < 32; ++d) out[d] = (_Float16)(ex[d] * inv);
  }
}

// ---------------------------------------------------------------------------
// attend1: y1o = 0.5*(q3@ctx1 + u1@ctx3) + x1 (fp32). One wave per (b,h,nt);
// both e-tiles per wave (A fragments loaded once, 4 WMMAs).
// ---------------------------------------------------------------------------
__global__ void attend1_kernel(const _Float16* __restrict__ P3y,
                               const _Float16* __restrict__ P1u,
                               const _Float16* __restrict__ Ctx1T,
                               const _Float16* __restrict__ Ctx3T,
                               const float* __restrict__ X1,
                               float* __restrict__ Y1O) {
  const int lane = threadIdx.x & 31;
  const int wave = threadIdx.x >> 5;
  const int idx = blockIdx.x * 8 + wave;
  const int nt = idx & 63;
  const int h  = (idx >> 6) & 7;
  const int b  = idx >> 9;
  const int m = lane & 15, kh = lane >> 4;
  const size_t arow  = (size_t)(b * N_ + nt * 16 + m) * C2_ + h * 32 + kh * 16;
  const size_t bbase = (size_t)(b * H_ + h) * 1024 + m * 32 + kh * 16;
  v16h a1 = load_frag_h(P3y + arow);
  v16h a2 = load_frag_h(P1u + arow);
  v8f c0 = {}, c1 = {};
  c0 = wmma_f16(a1, *(const v16h*)(Ctx1T + bbase), c0);
  c1 = wmma_f16(a1, *(const v16h*)(Ctx1T + bbase + 512), c1);
  c0 = wmma_f16(a2, *(const v16h*)(Ctx3T + bbase), c0);
  c1 = wmma_f16(a2, *(const v16h*)(Ctx3T + bbase + 512), c1);
  const int rbase = b * N_ + nt * 16 + 8 * kh;
  const int col0 = h * 32 + m;
#pragma unroll
  for (int r = 0; r < 8; ++r) {
    size_t g0 = (size_t)(rbase + r) * C_ + col0;
    Y1O[g0]      = 0.5f * c0[r] + X1[g0];
    Y1O[g0 + 16] = 0.5f * c1[r] + X1[g0 + 16];
  }
}

// ---------------------------------------------------------------------------
// attend2: y2o = concat(q3@ctx2, u2@ctx3) (f16 [BN,512]); both e-tiles per
// wave, `which` selects half.
// ---------------------------------------------------------------------------
__global__ void attend2_kernel(const _Float16* __restrict__ P3y,
                               const _Float16* __restrict__ P2u,
                               const _Float16* __restrict__ Ctx2T,
                               const _Float16* __restrict__ Ctx3T,
                               _Float16* __restrict__ Y2O) {
  const int lane = threadIdx.x & 31;
  const int wave = threadIdx.x >> 5;
  const int idx = blockIdx.x * 8 + wave;
  const int which = idx & 1;
  const int nt = (idx >> 1) & 63;
  const int h  = (idx >> 7) & 7;
  const int b  = idx >> 10;
  const int m = lane & 15, kh = lane >> 4;
  const _Float16* Ab = which ? P2u : P3y;
  const _Float16* Cb = which ? Ctx3T : Ctx2T;
  const size_t arow  = (size_t)(b * N_ + nt * 16 + m) * C2_ + h * 32 + kh * 16;
  const size_t bbase = (size_t)(b * H_ + h) * 1024 + m * 32 + kh * 16;
  v16h a = load_frag_h(Ab + arow);
  v8f c0 = {}, c1 = {};
  c0 = wmma_f16(a, *(const v16h*)(Cb + bbase), c0);
  c1 = wmma_f16(a, *(const v16h*)(Cb + bbase + 512), c1);
  const int col0  = (which ? C_ : 0) + h * 32 + m;
  const int rbase = b * N_ + nt * 16 + 8 * kh;
#pragma unroll
  for (int r = 0; r < 8; ++r) {
    size_t g0 = (size_t)(rbase + r) * C2_ + col0;
    Y2O[g0]      = (_Float16)c0[r];
    Y2O[g0 + 16] = (_Float16)c1[r];
  }
}

// ---------------------------------------------------------------------------
// end1: t[b,c,cls] = sum_n y1o[b,n,c]*We1[n,cls] + be1; layernorm over cls;
// out_x1[b,cls,c]. One block per b, thread = channel c; We1 staged in LDS.
// ---------------------------------------------------------------------------
__global__ void end1_kernel(const float* __restrict__ Y1O,
                            const float* __restrict__ We1,
                            const float* __restrict__ be1,
                            const float* __restrict__ g1,
                            const float* __restrict__ b1,
                            float* __restrict__ Out) {
  __shared__ float sW[N_ * CLS_];
  const int b = blockIdx.x;
  const int cidx = threadIdx.x;
  for (int i = cidx; i < N_ * CLS_; i += 256) sW[i] = We1[i];
  __syncthreads();
  float acc[CLS_] = {0.f, 0.f, 0.f, 0.f, 0.f, 0.f};
  const float* yp = Y1O + (size_t)b * N_ * C_ + cidx;
  for (int n = 0; n < N_; ++n) {
    float y = yp[(size_t)n * C_];
    const float* w = sW + n * CLS_;
#pragma unroll
    for (int j = 0; j < CLS_; ++j) acc[j] += y * w[j];
  }
  float mean = 0.f;
#pragma unroll
  for (int j = 0; j < CLS_; ++j) { acc[j] += be1[j]; mean += acc[j]; }
  mean *= (1.f / CLS_);
  float var = 0.f;
#pragma unroll
  for (int j = 0; j < CLS_; ++j) { float d = acc[j] - mean; var += d * d; }
  var *= (1.f / CLS_);
  float inv = rsqrtf(var + EPS_);
#pragma unroll
  for (int j = 0; j < CLS_; ++j)
    Out[((size_t)b * CLS_ + j) * C_ + cidx] = (acc[j] - mean) * inv * g1[j] + b1[j];
}

// ---------------------------------------------------------------------------
// end2: out_x2 = LN_256( x2 + y2o@We2 + be2 ). Block = 8 waves (256 thr), each
// wave a 16x32 stripe (K=512 -> 32 WMMAs, A loaded once); row LN fused in LDS.
// ---------------------------------------------------------------------------
__global__ void end2_kernel(const _Float16* __restrict__ Y2O,
                            const _Float16* __restrict__ We2t,
                            const float* __restrict__ be2,
                            const float* __restrict__ X2,
                            const float* __restrict__ g2,
                            const float* __restrict__ b2,
                            float* __restrict__ Out) {
  __shared__ float S[16][264];
  __shared__ float P[16][16], Q[16][16];
  __shared__ float Mn[16], Iv[16];
  const int lane = threadIdx.x & 31;
  const int wave = threadIdx.x >> 5;   // 0..7
  const int row0 = blockIdx.x * 16;
  const int col0 = wave * 32;
  const int m = lane & 15, kh = lane >> 4;
  const _Float16* arow = Y2O  + (size_t)(row0 + m) * C2_ + kh * 16;
  const _Float16* b0p  = We2t + (size_t)(col0 + m) * C2_ + kh * 16;
  const _Float16* b1p  = b0p + (size_t)16 * C2_;
  v8f c0 = {}, c1 = {};
#pragma unroll
  for (int kb = 0; kb < C2_; kb += 32) {
    v16h a = load_frag_h(arow + kb);
    c0 = wmma_f16(a, *(const v16h*)(b0p + kb), c0);
    c1 = wmma_f16(a, *(const v16h*)(b1p + kb), c1);
  }
  {
    const int col = col0 + m;
    const float bb = be2[col];
#pragma unroll
    for (int r = 0; r < 8; ++r) {
      int rr = 8 * kh + r;
      S[rr][col] = c0[r] + bb + X2[(size_t)(row0 + rr) * C_ + col];
    }
  }
  {
    const int col = col0 + 16 + m;
    const float bb = be2[col];
#pragma unroll
    for (int r = 0; r < 8; ++r) {
      int rr = 8 * kh + r;
      S[rr][col] = c1[r] + bb + X2[(size_t)(row0 + rr) * C_ + col];
    }
  }
  __syncthreads();
  {
    const int row = threadIdx.x >> 4;   // 0..15
    const int j = threadIdx.x & 15;
    float s = 0.f, sq = 0.f;
#pragma unroll
    for (int i = 0; i < 16; ++i) { float x = S[row][j * 16 + i]; s += x; sq += x * x; }
    P[row][j] = s; Q[row][j] = sq;
  }
  __syncthreads();
  if (threadIdx.x < 16) {
    float s = 0.f, sq = 0.f;
    for (int i = 0; i < 16; ++i) { s += P[threadIdx.x][i]; sq += Q[threadIdx.x][i]; }
    float mn = s * (1.f / 256.f);
    Mn[threadIdx.x] = mn;
    Iv[threadIdx.x] = rsqrtf(sq * (1.f / 256.f) - mn * mn + EPS_);
  }
  __syncthreads();
#pragma unroll
  for (int i = 0; i < 16; ++i) {
    int e = threadIdx.x + i * 256;
    int rr = e >> 8, cc = e & 255;
    Out[(size_t)(row0 + rr) * C_ + cc] = (S[rr][cc] - Mn[rr]) * Iv[rr] * g2[cc] + b2[cc];
  }
}

// ---------------------------------------------------------------------------
extern "C" void kernel_launch(void* const* d_in, const int* in_sizes, int n_in,
                              void* d_out, int out_size, void* d_ws, size_t ws_size,
                              hipStream_t stream) {
  (void)in_sizes; (void)n_in; (void)out_size; (void)ws_size;
  const float* x1  = (const float*)d_in[0];
  const float* x2  = (const float*)d_in[1];
  const float* seg = (const float*)d_in[2];
  const float* Wp1 = (const float*)d_in[3];  const float* bp1 = (const float*)d_in[4];
  const float* Wp2 = (const float*)d_in[5];  const float* bp2 = (const float*)d_in[6];
  const float* Wp3 = (const float*)d_in[7];  const float* bp3 = (const float*)d_in[8];
  const float* Wkv3 = (const float*)d_in[9];
  const float* Wkv1 = (const float*)d_in[10];
  const float* Wkv2 = (const float*)d_in[11];
  const float* We1 = (const float*)d_in[12]; const float* be1 = (const float*)d_in[13];
  const float* We2 = (const float*)d_in[14]; const float* be2 = (const float*)d_in[15];
  const float* g1  = (const float*)d_in[16]; const float* b1  = (const float*)d_in[17];
  const float* g2  = (const float*)d_in[18]; const float* b2  = (const float*)d_in[19];

  char* ws = (char*)d_ws;
  const size_t WT_BYTES = (size_t)C_ * C2_ * sizeof(_Float16);   // 256 KB each
  _Float16* WtP1  = (_Float16*)(ws + 0 * WT_BYTES);
  _Float16* WtP2  = (_Float16*)(ws + 1 * WT_BYTES);
  _Float16* WtP3  = (_Float16*)(ws + 2 * WT_BYTES);
  _Float16* WtKV1 = (_Float16*)(ws + 3 * WT_BYTES);
  _Float16* WtKV2 = (_Float16*)(ws + 4 * WT_BYTES);
  _Float16* WtKV3 = (_Float16*)(ws + 5 * WT_BYTES);
  _Float16* WtE2  = (_Float16*)(ws + 6 * WT_BYTES);
  _Float16* P1  = (_Float16*)(ws + 7 * WT_BYTES);   // f16 [BN,512] y|u
  _Float16* P2  = P1 + (size_t)BN_ * C2_;
  _Float16* P3  = P2 + (size_t)BN_ * C2_;
  _Float16* KV  = P3 + (size_t)BN_ * C2_;           // f16 [BN,512], reused per ctx
  _Float16* Y2O = KV;                               // alias: KV dead after ctx kernels
  _Float16* CT3 = KV + (size_t)BN_ * C2_;           // f16 ctxT [B,H,32e,32d]
  _Float16* CT1 = CT3 + (size_t)B_ * H_ * 32 * 32;
  _Float16* CT2 = CT1 + (size_t)B_ * H_ * 32 * 32;
  float*    Y1O = (float*)(CT2 + (size_t)B_ * H_ * 32 * 32);   // fp32 [BN,256]

  float* out1 = (float*)d_out;                   // [B, CLS, C]
  float* out2 = out1 + (size_t)B_ * CLS_ * C_;   // [B, N, C]

  const int TW = 256;
  const int NW = (C_ * C2_ + TW - 1) / TW;
  transpose_w_kernel<<<NW, TW, 0, stream>>>(Wp1,  WtP1,  C_,  C2_);
  transpose_w_kernel<<<NW, TW, 0, stream>>>(Wp2,  WtP2,  C_,  C2_);
  transpose_w_kernel<<<NW, TW, 0, stream>>>(Wp3,  WtP3,  C_,  C2_);
  transpose_w_kernel<<<NW, TW, 0, stream>>>(Wkv1, WtKV1, C_,  C2_);
  transpose_w_kernel<<<NW, TW, 0, stream>>>(Wkv2, WtKV2, C_,  C2_);
  transpose_w_kernel<<<NW, TW, 0, stream>>>(Wkv3, WtKV3, C_,  C2_);
  transpose_w_kernel<<<NW, TW, 0, stream>>>(We2,  WtE2,  C2_, C_);

  // (BN/32)*(512/32) 32x32-tiles, 8 waves per block
  const int GEMM_BLOCKS = (BN_ / 32) * (C2_ / 32) / 8;   // 2048
  proj_gemm_kernel<<<GEMM_BLOCKS, 256, 0, stream>>>(x1,  WtP1, bp1, P1);
  proj_gemm_kernel<<<GEMM_BLOCKS, 256, 0, stream>>>(x2,  WtP2, bp2, P2);
  proj_gemm_kernel<<<GEMM_BLOCKS, 256, 0, stream>>>(seg, WtP3, bp3, P3);

  // ctx3 from u3 (cols 256..511 of P3); ctx1 from y1; ctx2 from y2
  kv_gemm_kernel<<<GEMM_BLOCKS, 256, 0, stream>>>(P3 + C_, WtKV3, KV);
  ctx_kernel<<<B_ * H_, 128, 0, stream>>>(KV, CT3);
  kv_gemm_kernel<<<GEMM_BLOCKS, 256, 0, stream>>>(P1, WtKV1, KV);
  ctx_kernel<<<B_ * H_, 128, 0, stream>>>(KV, CT1);
  kv_gemm_kernel<<<GEMM_BLOCKS, 256, 0, stream>>>(P2, WtKV2, KV);
  ctx_kernel<<<B_ * H_, 128, 0, stream>>>(KV, CT2);

  attend1_kernel<<<(B_ * H_ * 64) / 8, 256, 0, stream>>>(P3, P1 + C_, CT1, CT3, x1, Y1O);
  attend2_kernel<<<(B_ * H_ * 64 * 2) / 8, 256, 0, stream>>>(P3, P2 + C_, CT2, CT3, Y2O);

  end1_kernel<<<B_, 256, 0, stream>>>(Y1O, We1, be1, g1, b1, out1);
  end2_kernel<<<BN_ / 16, 256, 0, stream>>>(Y2O, WtE2, be2, x2, g2, b2, out2);
}
